// Net_70643622085086
// MI455X (gfx1250) — matrix operand order repaired and checked by reading, not compiled, not run on hardware
//
#include <hip/hip_runtime.h>
#include <hip/hip_bf16.h>

#define NN   20000      // nodes
#define NE   320000     // edges
#define NG   512        // graphs
#define DM   256        // hidden dim
#define KHOP 4

typedef __attribute__((ext_vector_type(16))) __bf16 v16bf;
typedef __attribute__((ext_vector_type(8)))  __bf16 v8bf;
typedef __attribute__((ext_vector_type(8)))  float  v8f;

union Frag16 { v16bf v; v8bf h[2]; };

static inline int cdiv(long a, int b) { return (int)((a + (long)b - 1) / b); }

// ---------------- sparse / gather side ----------------

__global__ void k_embed(const int* __restrict__ x, const float* __restrict__ tbl,
                        float* __restrict__ h, int total /* NN*DM/4 */) {
    int t = blockIdx.x * blockDim.x + threadIdx.x;
    if (t >= total) return;
    int row = t >> 6;            // DM/4 == 64
    int c4  = t & 63;
    ((float4*)h)[t] = ((const float4*)tbl)[x[row] * 64 + c4];
}

__global__ void k_fill(float* __restrict__ p, float v, int n) {
    int t = blockIdx.x * blockDim.x + threadIdx.x;
    if (t < n) p[t] = v;
}

__global__ void k_deg_acc(const int* __restrict__ dst, float* __restrict__ deg, int e) {
    int t = blockIdx.x * blockDim.x + threadIdx.x;
    if (t < e) atomicAdd(&deg[dst[t]], 1.0f);
}

__global__ void k_rsqrt_inplace(float* __restrict__ p, int n) {
    int t = blockIdx.x * blockDim.x + threadIdx.x;
    if (t < n) p[t] = rsqrtf(p[t]);   // deg >= 1 (self loop), so always valid
}

// self-loop term initializes the destination buffer: hn = h * dinv^2
__global__ void k_hop_init(const float* __restrict__ h, const float* __restrict__ dinv,
                           float* __restrict__ hn, int total /* NN*64 */) {
    int t = blockIdx.x * blockDim.x + threadIdx.x;
    if (t >= total) return;
    int i = t >> 6;
    float w = dinv[i] * dinv[i];
    float4 v = ((const float4*)h)[t];
    v.x *= w; v.y *= w; v.z *= w; v.w *= w;
    ((float4*)hn)[t] = v;
}

__global__ void k_scatter(const int* __restrict__ src, const int* __restrict__ dst,
                          const float* __restrict__ dinv, const float* __restrict__ h,
                          float* __restrict__ hn, int total /* NE*64 */) {
    int t = blockIdx.x * blockDim.x + threadIdx.x;
    if (t >= total) return;
    int e = t >> 6;
    int c = (t & 63) << 2;
    int s = src[e], d = dst[e];
    float w = dinv[s] * dinv[d];
    const float* hp = h  + (long)s * DM + c;
    float*       op = hn + (long)d * DM + c;
    float4 v = *(const float4*)hp;
    atomicAdd(op + 0, v.x * w);
    atomicAdd(op + 1, v.y * w);
    atomicAdd(op + 2, v.z * w);
    atomicAdd(op + 3, v.w * w);
}

__global__ void k_count(const int* __restrict__ batch, float* __restrict__ cnt, int n) {
    int t = blockIdx.x * blockDim.x + threadIdx.x;
    if (t < n) atomicAdd(&cnt[batch[t]], 1.0f);
}

__global__ void k_pool(const int* __restrict__ batch, const float* __restrict__ h,
                       float* __restrict__ hg, int total /* NN*64 */) {
    int t = blockIdx.x * blockDim.x + threadIdx.x;
    if (t >= total) return;
    int i = t >> 6;
    int c = (t & 63) << 2;
    int g = batch[i];
    float4 v = *(const float4*)(h + (long)i * DM + c);
    float* op = hg + (long)g * DM + c;
    atomicAdd(op + 0, v.x);
    atomicAdd(op + 1, v.y);
    atomicAdd(op + 2, v.z);
    atomicAdd(op + 3, v.w);
}

// ---------------- dense side (WMMA bf16) ----------------

__global__ void k_to_bf16(const float* __restrict__ s, __bf16* __restrict__ d, int n) {
    int t = blockIdx.x * blockDim.x + threadIdx.x;
    if (t < n) d[t] = (__bf16)s[t];
}

// convert + transpose: src is [K,N] row-major f32, dst is [N,K] row-major bf16
__global__ void k_to_bf16_T(const float* __restrict__ s, __bf16* __restrict__ d,
                            int K, int N) {
    int t = blockIdx.x * blockDim.x + threadIdx.x;
    if (t >= K * N) return;
    int k = t / N, n = t - k * N;
    d[(long)n * K + k] = (__bf16)s[t];
}

// C[M,N] = A[M,K] @ Bt[N,K]^T + rowscale[i]*bias[j]
// One wave per 16x16 C tile; K is a compile-time constant (multiple of 32).
// Fragment packing per CDNA5 ISA 7.12.2 (wave32):
//   A 16x32 bf16: lane row = lane&15; elems 0..7  -> K = half*8 + e      (contiguous -> b128)
//                                     elems 8..15 -> K = 16 + half*8 + e (contiguous -> b128)
//   B 32x16 bf16: lane col = lane&15; same K mapping as A                (contiguous in Bt -> 2x b128)
//   C/D 16x16 f32: vgpr r -> row = half*8 + r, col = lane&15
template <int K>
__global__ void k_gemm_wmma(const __bf16* __restrict__ A, const __bf16* __restrict__ Bt,
                            float* __restrict__ C, int M, int N,
                            const float* __restrict__ bias,
                            const float* __restrict__ rowscale) {
    const int tilesN = N >> 4;
    const int tm = (blockIdx.x / tilesN) << 4;
    const int tn = (blockIdx.x % tilesN) << 4;
    const int lane = threadIdx.x;
    const int half = lane >> 4;
    const int l15  = lane & 15;

    const __bf16* ap = A  + (long)(tm + l15) * K + half * 8;   // K-contiguous
    const __bf16* bp = Bt + (long)(tn + l15) * K + half * 8;   // K-contiguous

    v8f acc = {};
    #pragma unroll
    for (int kk = 0; kk < K; kk += 32) {
        Frag16 a, b;
        a.h[0] = *(const v8bf*)(ap + kk);
        a.h[1] = *(const v8bf*)(ap + kk + 16);
        b.h[0] = *(const v8bf*)(bp + kk);
        b.h[1] = *(const v8bf*)(bp + kk + 16);
        acc = __builtin_amdgcn_wmma_f32_16x16x32_bf16(
            /*neg_a=*/false, a.v, /*neg_b=*/false, b.v,
            /*c_mod=*/(short)0, acc, /*reuse_a=*/false, /*reuse_b=*/false);
    }

    const int col = tn + l15;
    const float bj = bias ? bias[col] : 0.0f;
    #pragma unroll
    for (int r = 0; r < 8; ++r) {
        const int row = tm + half * 8 + r;
        const float s = rowscale ? rowscale[row] : 1.0f;
        C[(long)row * N + col] = acc[r] + s * bj;
    }
}

__global__ void k_bn_stats(const float* __restrict__ z, float* __restrict__ mean,
                           float* __restrict__ var, int M, int N) {
    int j = blockIdx.x * blockDim.x + threadIdx.x;
    if (j >= N) return;
    float s = 0.f, s2 = 0.f;
    for (int i = 0; i < M; ++i) {
        float v = z[(long)i * N + j];
        s += v; s2 += v * v;
    }
    float m = s / (float)M;
    mean[j] = m;
    var[j]  = s2 / (float)M - m * m;
}

__global__ void k_bn_relu(float* __restrict__ z, const float* __restrict__ mean,
                          const float* __restrict__ var, const float* __restrict__ gamma,
                          const float* __restrict__ beta, int M, int N) {
    int t = blockIdx.x * blockDim.x + threadIdx.x;
    if (t >= M * N) return;
    int j = t % N;
    float v = (z[t] - mean[j]) * rsqrtf(var[j] + 1e-5f) * gamma[j] + beta[j];
    z[t] = v > 0.f ? v : 0.f;
}

// out[g] = dot(a[g,:], w) + b  (one wave per row, wave32 shuffle reduction)
__global__ void k_final(const float* __restrict__ a, const float* __restrict__ w,
                        const float* __restrict__ b, float* __restrict__ out, int D_) {
    int g = blockIdx.x;
    int lane = threadIdx.x;
    float s = 0.f;
    for (int d = lane; d < D_; d += 32) s += a[(long)g * D_ + d] * w[d];
    #pragma unroll
    for (int off = 16; off > 0; off >>= 1) s += __shfl_down(s, off, 32);
    if (lane == 0) out[g] = s + b[0];
}

// ---------------- launch ----------------

extern "C" void kernel_launch(void* const* d_in, const int* in_sizes, int n_in,
                              void* d_out, int out_size, void* d_ws, size_t ws_size,
                              hipStream_t stream) {
    const int*   x      = (const int*)d_in[0];
    const int*   ei     = (const int*)d_in[1];          // [2, NE]
    const int*   batch  = (const int*)d_in[2];
    const float* tbl    = (const float*)d_in[3];
    const float* sg_W   = (const float*)d_in[4];
    const float* sg_b   = (const float*)d_in[5];
    const float* W1     = (const float*)d_in[6];
    const float* b1     = (const float*)d_in[7];
    const float* g1     = (const float*)d_in[8];
    const float* be1    = (const float*)d_in[9];
    const float* W2     = (const float*)d_in[10];
    const float* b2     = (const float*)d_in[11];
    const float* g2     = (const float*)d_in[12];
    const float* be2    = (const float*)d_in[13];
    const float* W3     = (const float*)d_in[14];
    const float* b3     = (const float*)d_in[15];
    float* out = (float*)d_out;

    const int* esrc = ei;
    const int* edst = ei + NE;

    // workspace carve-up (float units, 64-float aligned blocks)
    float* w = (float*)d_ws;
    float* h0   = w;  w += (long)NN * DM;       // 5,120,000
    float* h1   = w;  w += (long)NN * DM;       // 5,120,000
    float* dinv = w;  w += 20032;
    float* cnt  = w;  w += NG;
    float* hg   = w;  w += NG * DM;             // pooled
    float* t1   = w;  w += NG * DM;             // hg @ sg_W (+ cnt*sg_b)
    float* z1   = w;  w += NG * 2 * DM;
    float* z2   = w;  w += NG * DM;
    float* mean = w;  w += 512;
    float* var  = w;  w += 512;
    __bf16* bfA = (__bf16*)w;  w += (NG * 2 * DM) / 2;   // up to 262144 bf16
    __bf16* bfW = (__bf16*)w;  w += (2 * DM * DM) / 2;   // up to 131072 bf16 (transposed)

    const int TB = 256;

    // 1) embedding gather
    k_embed<<<cdiv((long)NN * 64, TB), TB, 0, stream>>>(x, tbl, h0, NN * 64);

    // 2) degree (self loop => init to 1), then dinv = rsqrt(deg) in place
    k_fill<<<cdiv(NN, TB), TB, 0, stream>>>(dinv, 1.0f, NN);
    k_deg_acc<<<cdiv(NE, TB), TB, 0, stream>>>(edst, dinv, NE);
    k_rsqrt_inplace<<<cdiv(NN, TB), TB, 0, stream>>>(dinv, NN);

    // 3) K hops: hn = h*dinv^2 (self loop) then scatter-add edges
    float* hcur = h0;
    float* hnext = h1;
    for (int hop = 0; hop < KHOP; ++hop) {
        k_hop_init<<<cdiv((long)NN * 64, TB), TB, 0, stream>>>(hcur, dinv, hnext, NN * 64);
        k_scatter<<<cdiv((long)NE * 64, TB), TB, 0, stream>>>(esrc, edst, dinv, hcur, hnext, NE * 64);
        float* t = hcur; hcur = hnext; hnext = t;
    }

    // 4) pool first (linear => commutes with sg_W), plus per-graph node counts for sg_b
    k_fill<<<cdiv(NG * DM, TB), TB, 0, stream>>>(hg, 0.0f, NG * DM);
    k_fill<<<cdiv(NG, TB), TB, 0, stream>>>(cnt, 0.0f, NG);
    k_count<<<cdiv(NN, TB), TB, 0, stream>>>(batch, cnt, NN);
    k_pool<<<cdiv((long)NN * 64, TB), TB, 0, stream>>>(batch, hcur, hg, NN * 64);

    // 5) t1 = hg @ sg_W + cnt[g]*sg_b   (512x256x256, WMMA bf16, fused bias)
    k_to_bf16<<<cdiv(NG * DM, TB), TB, 0, stream>>>(hg, bfA, NG * DM);
    k_to_bf16_T<<<cdiv(DM * DM, TB), TB, 0, stream>>>(sg_W, bfW, DM, DM);
    k_gemm_wmma<DM><<<(NG / 16) * (DM / 16), 32, 0, stream>>>(
        bfA, bfW, t1, NG, DM, sg_b, cnt);

    // 6) z1 = relu(BN(t1 @ W1 + b1))    (512x512x256)
    k_to_bf16<<<cdiv(NG * DM, TB), TB, 0, stream>>>(t1, bfA, NG * DM);
    k_to_bf16_T<<<cdiv(DM * 2 * DM, TB), TB, 0, stream>>>(W1, bfW, DM, 2 * DM);
    k_gemm_wmma<DM><<<(NG / 16) * (2 * DM / 16), 32, 0, stream>>>(
        bfA, bfW, z1, NG, 2 * DM, b1, nullptr);
    k_bn_stats<<<cdiv(2 * DM, TB), TB, 0, stream>>>(z1, mean, var, NG, 2 * DM);
    k_bn_relu<<<cdiv(NG * 2 * DM, TB), TB, 0, stream>>>(z1, mean, var, g1, be1, NG, 2 * DM);

    // 7) z2 = relu(BN(z1 @ W2 + b2))    (512x256x512)
    k_to_bf16<<<cdiv(NG * 2 * DM, TB), TB, 0, stream>>>(z1, bfA, NG * 2 * DM);
    k_to_bf16_T<<<cdiv(2 * DM * DM, TB), TB, 0, stream>>>(W2, bfW, 2 * DM, DM);
    k_gemm_wmma<2 * DM><<<(NG / 16) * (DM / 16), 32, 0, stream>>>(
        bfA, bfW, z2, NG, DM, b2, nullptr);
    k_bn_stats<<<cdiv(DM, TB), TB, 0, stream>>>(z2, mean, var, NG, DM);
    k_bn_relu<<<cdiv(NG * DM, TB), TB, 0, stream>>>(z2, mean, var, g2, be2, NG, DM);

    // 8) out = z2 @ W3 + b3
    k_final<<<NG, 32, 0, stream>>>(z2, W3, b3, out, DM);
}
